// CrossWinAttention_81346680586407
// MI455X (gfx1250) — matrix operand order repaired and checked by reading, not compiled, not run on hardware
//
#include <hip/hip_runtime.h>
#include <hip/hip_bf16.h>

typedef __attribute__((ext_vector_type(16))) _Float16 v16h;
typedef __attribute__((ext_vector_type(8)))  _Float16 v8h;
typedef __attribute__((ext_vector_type(8)))  float    v8f;

// Problem constants
#define BB_   2
#define NN_   6
#define LL_   64     // X*Y
#define WW_   64     // W1*W2
#define DD_   128
#define HH_   4
#define DHH_  32
#define QT_   384    // tokens per window = N*W1*W2
#define NWAVE 8
#define EPSLN 1e-5f

// LDS layout (bytes)
#define OFF_X   0                          // f16 [384][128]  = 98304
#define OFF_Q   (OFF_X + QT_*DD_*2)        // f16 [384][32]   = 24576
#define OFF_K   (OFF_Q + QT_*DHH_*2)       // f16 [384][32]   = 24576
#define OFF_VT  (OFF_K + QT_*DHH_*2)       // f16 [32][384]   = 24576
#define OFF_WT  (OFF_VT + DHH_*QT_*2)      // f16 [32][128]   = 8192
#define OFF_WPT (OFF_WT + DHH_*DD_*2)      // f16 [128][32]   = 8192
#define OFF_P   (OFF_WPT + DD_*DHH_*2)     // f16 [8 waves][16][32] = 8192
#define OFF_Z   (OFF_P + NWAVE*16*32*2)    // f32 [64][128]   = 32768
#define SMEM_BYTES (OFF_Z + WW_*DD_*4)     // = 229376

static __device__ __forceinline__ v8f wmma_f16(v16h a, v16h b, v8f c) {
    return __builtin_amdgcn_wmma_f32_16x16x32_f16(false, a, false, b, (short)0, c, false, false);
}

// A operand (16x32 f16, M rows x K): lane<16 -> row=m0+lane, K = k0+[0..7, 16..23]
//                                    lane>=16 -> row=m0+lane-16, K = k0+[8..15, 24..31]
static __device__ __forceinline__ v16h load_a16(const _Float16* base, int strideH,
                                                int m0, int k0, int lane) {
    int hf = lane >> 4;
    const _Float16* p = base + (m0 + (lane & 15)) * strideH + k0 + hf * 8;
    v8h lo = *(const v8h*)p;
    v8h hi = *(const v8h*)(p + 16);
    return __builtin_shufflevector(lo, hi, 0,1,2,3,4,5,6,7,8,9,10,11,12,13,14,15);
}

// B operand (32x16 f16, K x N) from K-major storage base[ncol][k]:
// lane<16 -> col=n0+lane, K = k0+[0..7, 16..23]; lane>=16 -> col=n0+lane-16, K = k0+[8..15, 24..31]
static __device__ __forceinline__ v16h load_b16(const _Float16* base, int strideH,
                                                int k0, int n0, int lane) {
    int hf = lane >> 4;
    const _Float16* p = base + (n0 + (lane & 15)) * strideH + k0 + hf * 8;
    v8h lo = *(const v8h*)p;
    v8h hi = *(const v8h*)(p + 16);
    return __builtin_shufflevector(lo, hi, 0,1,2,3,4,5,6,7,8,9,10,11,12,13,14,15);
}

static __device__ __forceinline__ float half16_max(float v) {
    v = fmaxf(v, __shfl_xor(v, 1, 32));
    v = fmaxf(v, __shfl_xor(v, 2, 32));
    v = fmaxf(v, __shfl_xor(v, 4, 32));
    v = fmaxf(v, __shfl_xor(v, 8, 32));
    return v;
}
static __device__ __forceinline__ float half16_sum(float v) {
    v += __shfl_xor(v, 1, 32);
    v += __shfl_xor(v, 2, 32);
    v += __shfl_xor(v, 4, 32);
    v += __shfl_xor(v, 8, 32);
    return v;
}
static __device__ __forceinline__ float wave32_sum(float v) {
    v += __shfl_xor(v, 1, 32);
    v += __shfl_xor(v, 2, 32);
    v += __shfl_xor(v, 4, 32);
    v += __shfl_xor(v, 8, 32);
    v += __shfl_xor(v, 16, 32);
    return v;
}

// TSEL: 0 -> store row-major Q, 1 -> store row-major K, 2 -> store transposed V.
// Compile-time TSEL keeps every ds_store unconditional (no per-store branching).
template<int TSEL>
static __device__ __forceinline__ void project_tensor(
    const float* __restrict__ src, const float* __restrict__ gg, const float* __restrict__ be,
    const float* __restrict__ W, const float* __restrict__ bias,
    _Float16* __restrict__ Xl, _Float16* __restrict__ WtT,
    _Float16* __restrict__ dQ, _Float16* __restrict__ dK, _Float16* __restrict__ dVt,
    int b, int l, int h, float fac,
    int tid, int lane, int w, int hf, int ln16, v8f zro)
{
    // ---- LayerNorm 384 tokens -> f16 X (one token per wave iteration) ----
    for (int i = 0; i < QT_ / NWAVE; ++i) {
        int tok = w * (QT_ / NWAVE) + i;
        int n   = tok >> 6;
        int ww2 = tok & 63;
        const float* p = src + ((((long)(b * NN_ + n)) * LL_ + l) * WW_ + ww2) * DD_ + lane * 4;
        float4 x4 = *(const float4*)p;
        float s  = x4.x + x4.y + x4.z + x4.w;
        float s2 = x4.x * x4.x + x4.y * x4.y + x4.z * x4.z + x4.w * x4.w;
        s  = wave32_sum(s);
        s2 = wave32_sum(s2);
        float mean = s * (1.0f / 128.0f);
        float var  = s2 * (1.0f / 128.0f) - mean * mean;
        float rs   = rsqrtf(var + EPSLN);
        int dbase = lane * 4;
        float4 g4 = *(const float4*)(gg + dbase);
        float4 b4 = *(const float4*)(be + dbase);
        _Float16* xp = Xl + tok * DD_ + dbase;
        xp[0] = (_Float16)((x4.x - mean) * rs * g4.x + b4.x);
        xp[1] = (_Float16)((x4.y - mean) * rs * g4.y + b4.y);
        xp[2] = (_Float16)((x4.z - mean) * rs * g4.z + b4.z);
        xp[3] = (_Float16)((x4.w - mean) * rs * g4.w + b4.w);
    }
    // ---- weight head slice transposed to [outcol 32][k 128] ----
    for (int i = tid; i < DHH_ * DD_; i += 256) {
        int col = i >> 7, kk = i & 127;
        WtT[col * DD_ + kk] = (_Float16)W[kk * DD_ + h * DHH_ + col];
    }
    __syncthreads();
    // ---- projection GEMM (384x128)x(128x32): 24 rowtiles x 2 ntiles ----
    for (int tt = 0; tt < 6; ++tt) {
        int tile = w * 6 + tt;
        int rt = tile >> 1, nt = tile & 1;
        v8f acc = zro;
#pragma unroll
        for (int ks = 0; ks < 4; ++ks) {
            v16h a  = load_a16(Xl, DD_, rt * 16, ks * 32, lane);
            v16h bm = load_b16(WtT, DD_, ks * 32, nt * 16, lane);
            acc = wmma_f16(a, bm, acc);
        }
        float bv = bias[h * DHH_ + nt * 16 + ln16];
        int m0 = rt * 16;
#pragma unroll
        for (int r = 0; r < 8; ++r) {
            float val = (acc[r] + bv) * fac;
            int row = m0 + r + hf * 8;
            int col = nt * 16 + ln16;
            if (TSEL == 0)      dQ[row * DHH_ + col] = (_Float16)val;
            else if (TSEL == 1) dK[row * DHH_ + col] = (_Float16)val;
            else                dVt[col * QT_ + row] = (_Float16)val;
        }
    }
}

__global__ void __launch_bounds__(256, 1)
cross_win_attn_kernel(const float* __restrict__ qg, const float* __restrict__ kg,
                      const float* __restrict__ vg, const float* __restrict__ skip,
                      const float* __restrict__ head_gate,
                      const float* __restrict__ lnq_g, const float* __restrict__ lnq_b,
                      const float* __restrict__ lnk_g, const float* __restrict__ lnk_b,
                      const float* __restrict__ lnv_g, const float* __restrict__ lnv_b,
                      const float* __restrict__ Wq, const float* __restrict__ bq,
                      const float* __restrict__ Wk, const float* __restrict__ bk,
                      const float* __restrict__ Wv, const float* __restrict__ bv,
                      const float* __restrict__ Wp, const float* __restrict__ bp,
                      float* __restrict__ out)
{
    extern __shared__ char smem[];
    _Float16* Xl  = (_Float16*)(smem + OFF_X);
    _Float16* Ql  = (_Float16*)(smem + OFF_Q);
    _Float16* Kl  = (_Float16*)(smem + OFF_K);
    _Float16* Vt  = (_Float16*)(smem + OFF_VT);
    _Float16* WtT = (_Float16*)(smem + OFF_WT);
    _Float16* WpT = (_Float16*)(smem + OFF_WPT);
    _Float16* Ps  = (_Float16*)(smem + OFF_P);
    float*    Zf  = (float*)(smem + OFF_Z);

    const int tid  = threadIdx.x;
    const int lane = tid & 31;
    const int w    = tid >> 5;
    const int hf   = lane >> 4;
    const int ln16 = lane & 15;
    const int bid  = blockIdx.x;
    const int b    = bid >> 6;      // window batch
    const int l    = bid & 63;      // window (x,y)

    for (int i = tid; i < WW_ * DD_; i += 256) Zf[i] = 0.0f;

    v8f Zacc[8];
#pragma unroll
    for (int i = 0; i < 8; ++i)
#pragma unroll
        for (int e = 0; e < 8; ++e) Zacc[i][e] = 0.0f;
    v8f zro;
#pragma unroll
    for (int e = 0; e < 8; ++e) zro[e] = 0.0f;

    _Float16* Pw = Ps + w * 16 * 32;

    for (int h = 0; h < HH_; ++h) {
        __syncthreads();
        // Wp head slice, transposed to [outcol 128][k 32]
        for (int i = tid; i < DD_ * DHH_; i += 256) {
            int col = i >> 5, kk = i & 31;
            WpT[col * DHH_ + kk] = (_Float16)Wp[(h * DHH_ + kk) * DD_ + col];
        }
        const float gh = head_gate[h];
        // warm GL2 with next head's weight slices (global_prefetch_b8)
        if (h + 1 < HH_) {
            __builtin_prefetch(Wp + (h + 1) * DHH_ * DD_ + tid * 16, 0, 1);
            __builtin_prefetch(Wq + (h + 1) * DHH_ + tid * DD_, 0, 1);
        }

        __syncthreads();
        project_tensor<0>(qg, lnq_g, lnq_b, Wq, bq, Xl, WtT, Ql, Kl, Vt,
                          b, l, h, gh * 0.17677669529663687f, tid, lane, w, hf, ln16, zro);
        __syncthreads();
        project_tensor<1>(kg, lnk_g, lnk_b, Wk, bk, Xl, WtT, Ql, Kl, Vt,
                          b, l, h, gh, tid, lane, w, hf, ln16, zro);
        __syncthreads();
        project_tensor<2>(vg, lnv_g, lnv_b, Wv, bv, Xl, WtT, Ql, Kl, Vt,
                          b, l, h, gh, tid, lane, w, hf, ln16, zro);
        __syncthreads();

        // ---- flash attention, 3 query tiles per wave ----
        for (int j = 0; j < 3; ++j) {
            int qt = w + 8 * j;
            int m0 = qt * 16;
            v16h aq = load_a16(Ql, DHH_, m0, 0, lane);
            float mrun[8], lrun[8];
            v8f o0 = zro, o1 = zro;
#pragma unroll
            for (int r = 0; r < 8; ++r) { mrun[r] = -3.0e38f; lrun[r] = 0.0f; }
            for (int kb = 0; kb < 12; ++kb) {
                v16h bk0 = load_b16(Kl, DHH_, 0, kb * 32, lane);
                v16h bk1 = load_b16(Kl, DHH_, 0, kb * 32 + 16, lane);
                v8f s0 = wmma_f16(aq, bk0, zro);
                v8f s1 = wmma_f16(aq, bk1, zro);
#pragma unroll
                for (int r = 0; r < 8; ++r) {
                    float bm = half16_max(fmaxf(s0[r], s1[r]));
                    float mn = fmaxf(mrun[r], bm);
                    float corr = __expf(mrun[r] - mn);
                    float p0 = __expf(s0[r] - mn);
                    float p1 = __expf(s1[r] - mn);
                    float rs = half16_sum(p0 + p1);
                    lrun[r] = lrun[r] * corr + rs;
                    mrun[r] = mn;
                    o0[r] *= corr;
                    o1[r] *= corr;
                    int row = r + hf * 8;
                    Pw[row * 32 + ln16]      = (_Float16)p0;
                    Pw[row * 32 + 16 + ln16] = (_Float16)p1;
                }
                v16h ap  = load_a16(Pw, 32, 0, 0, lane);
                v16h bv0 = load_b16(Vt, QT_, kb * 32, 0, lane);
                v16h bv1 = load_b16(Vt, QT_, kb * 32, 16, lane);
                o0 = wmma_f16(ap, bv0, o0);
                o1 = wmma_f16(ap, bv1, o1);
            }
            // normalize, convert O -> A layout via per-wave scratch
#pragma unroll
            for (int r = 0; r < 8; ++r) {
                float inv = 1.0f / lrun[r];
                int row = r + hf * 8;
                Pw[row * 32 + ln16]      = (_Float16)(o0[r] * inv);
                Pw[row * 32 + 16 + ln16] = (_Float16)(o1[r] * inv);
            }
            v16h ao = load_a16(Pw, 32, 0, 0, lane);
            // output projection partial; mean-over-n via non-reset accumulators
#pragma unroll
            for (int nt = 0; nt < 8; ++nt) {
                v16h bz = load_b16(WpT, DHH_, 0, nt * 16, lane);
                Zacc[nt] = wmma_f16(ao, bz, Zacc[nt]);
            }
        }
    }
    __syncthreads();
    // combine the two waves sharing each output row group
    int g = w & 3;
#pragma unroll
    for (int nt = 0; nt < 8; ++nt)
#pragma unroll
        for (int r = 0; r < 8; ++r) {
            int row = g * 16 + r + hf * 8;
            atomicAdd(&Zf[row * DD_ + nt * 16 + ln16], Zacc[nt][r]);
        }
    __syncthreads();
    const float* skipB = skip + (long)bid * WW_ * DD_;
    float*       outB  = out + (long)bid * WW_ * DD_;
    for (int i = tid; i < WW_ * DD_; i += 256) {
        int col = i & 127;
        outB[i] = Zf[i] * (1.0f / 6.0f) + bp[col] + skipB[i];
    }
}

extern "C" void kernel_launch(void* const* d_in, const int* in_sizes, int n_in,
                              void* d_out, int out_size, void* d_ws, size_t ws_size,
                              hipStream_t stream) {
    (void)in_sizes; (void)n_in; (void)out_size; (void)d_ws; (void)ws_size;
    const float* q         = (const float*)d_in[0];
    const float* k         = (const float*)d_in[1];
    const float* v         = (const float*)d_in[2];
    const float* skip      = (const float*)d_in[3];
    const float* head_gate = (const float*)d_in[4];
    const float* lnq_g     = (const float*)d_in[5];
    const float* lnq_b     = (const float*)d_in[6];
    const float* lnk_g     = (const float*)d_in[7];
    const float* lnk_b     = (const float*)d_in[8];
    const float* lnv_g     = (const float*)d_in[9];
    const float* lnv_b     = (const float*)d_in[10];
    const float* Wq        = (const float*)d_in[11];
    const float* bq        = (const float*)d_in[12];
    const float* Wk        = (const float*)d_in[13];
    const float* bk        = (const float*)d_in[14];
    const float* Wv        = (const float*)d_in[15];
    const float* bv        = (const float*)d_in[16];
    const float* Wp        = (const float*)d_in[17];
    const float* bp        = (const float*)d_in[18];
    float* out = (float*)d_out;

    hipFuncSetAttribute((const void*)cross_win_attn_kernel,
                        hipFuncAttributeMaxDynamicSharedMemorySize, SMEM_BYTES);
    cross_win_attn_kernel<<<dim3(BB_ * LL_), dim3(256), SMEM_BYTES, stream>>>(
        q, k, v, skip, head_gate, lnq_g, lnq_b, lnk_g, lnk_b, lnv_g, lnv_b,
        Wq, bq, Wk, bk, Wv, bv, Wp, bp, out);
}
